// CausalSelfAttention_7962869367571
// MI455X (gfx1250) — matrix operand order repaired and checked by reading, not compiled, not run on hardware
//
#include <hip/hip_runtime.h>
#include <hip/hip_bf16.h>

typedef __bf16 bf16_t;
typedef bf16_t v16bf __attribute__((ext_vector_type(16)));
typedef bf16_t v8bf  __attribute__((ext_vector_type(8)));
typedef bf16_t v4bf  __attribute__((ext_vector_type(4)));
typedef float  v8f   __attribute__((ext_vector_type(8)));

#define E_DIM 1024
#define H_DIM 16
#define D_DIM 64
#define B_DIM 2
#define T_DIM 2048
#define M_DIM (B_DIM * T_DIM)   /* 4096 rows of x */

#define WMMA_BF16(a, b, c) \
    __builtin_amdgcn_wmma_f32_16x16x32_bf16(false, (a), false, (b), (short)0, (c), false, false)

// wave32 xor-shuffle via ds_swizzle (group-of-32 encoding: [14:10]=xor, [9:5]=or, [4:0]=and)
#define SWZ_XOR_F(v, mask) __int_as_float(__builtin_amdgcn_ds_swizzle(__float_as_int(v), (((mask) << 10) | 0x1f)))

__device__ __forceinline__ float rowmax16(float v) {
    v = fmaxf(v, SWZ_XOR_F(v, 1));
    v = fmaxf(v, SWZ_XOR_F(v, 2));
    v = fmaxf(v, SWZ_XOR_F(v, 4));
    v = fmaxf(v, SWZ_XOR_F(v, 8));
    return v;
}
__device__ __forceinline__ float rowsum16(float v) {
    v += SWZ_XOR_F(v, 1);
    v += SWZ_XOR_F(v, 2);
    v += SWZ_XOR_F(v, 4);
    v += SWZ_XOR_F(v, 8);
    return v;
}

// Build a 16-bit A-fragment (16x32) for this lane from two 16-byte chunks.
__device__ __forceinline__ v16bf make_afrag(const bf16_t* p0, const bf16_t* p1) {
    v8bf lo = *(const v8bf*)p0;
    v8bf hi = *(const v8bf*)p1;
    return __builtin_shufflevector(lo, hi, 0, 1, 2, 3, 4, 5, 6, 7, 8, 9, 10, 11, 12, 13, 14, 15);
}

// ---------------- conversion kernels ----------------

__global__ void f32_to_bf16_x4_kernel(const float* __restrict__ in, bf16_t* __restrict__ out, int n4) {
    int i = blockIdx.x * blockDim.x + threadIdx.x;
    if (i < n4) {
        float4 v = ((const float4*)in)[i];
        v4bf o;
        o[0] = (bf16_t)v.x; o[1] = (bf16_t)v.y; o[2] = (bf16_t)v.z; o[3] = (bf16_t)v.w;
        ((v4bf*)out)[i] = o;
    }
}

// WT[n][k] = W[k][n], fp32 -> bf16 (E x E), LDS-tiled so both sides coalesce.
__global__ void transpose_w_bf16_kernel(const float* __restrict__ W, bf16_t* __restrict__ WT) {
    __shared__ float tile[32][33];
    int kb = blockIdx.x * 32;
    int nb = blockIdx.y * 32;
    int tx = threadIdx.x;          // 0..31
    int ty = threadIdx.y;          // 0..7
#pragma unroll
    for (int i = 0; i < 32; i += 8)
        tile[ty + i][tx] = W[(kb + ty + i) * E_DIM + nb + tx];   // tile[dk][dn]
    __syncthreads();
#pragma unroll
    for (int i = 0; i < 32; i += 8)
        WT[(nb + ty + i) * E_DIM + kb + tx] = (bf16_t)tile[tx][ty + i];
}

// ---------------- projection GEMM ----------------
// Y[m][n] = sum_k X[m][k] * WT[n][k] + bias[n]
// One wave computes a 64x64 output tile (16 WMMAs per 32-wide k-step).
// The K loop is fully unrolled: K is compile-time, so every load is
// base + immediate offset (no address VALU in the stream, no WMMA->VALU
// WAR hazard NOPs, no loop scalar overhead) and the ping-pong fragment
// sets alternate purely by register naming.
// MODE 0: bf16 out[B,H,T,D]   (Q, K)
// MODE 1: bf16 out[B,H,D,T]   (V transposed)
// MODE 2: f32  out[M,E]       (final projection)

__device__ __forceinline__ void gemm_load(const bf16_t* (&xr)[4], const bf16_t* (&wr)[4],
                                          int kb, int hi,
                                          v16bf (&a)[4], v16bf (&bm)[4]) {
#pragma unroll
    for (int mi = 0; mi < 4; ++mi)
        a[mi] = make_afrag(xr[mi] + kb + hi * 16, xr[mi] + kb + hi * 16 + 8);
#pragma unroll
    for (int nt = 0; nt < 4; ++nt)
        bm[nt] = *(const v16bf*)(wr[nt] + kb + hi * 16);
}

__device__ __forceinline__ void gemm_mma(const v16bf (&a)[4], const v16bf (&bm)[4], v8f (&acc)[16]) {
#pragma unroll
    for (int mi = 0; mi < 4; ++mi)
#pragma unroll
        for (int nt = 0; nt < 4; ++nt)
            acc[mi * 4 + nt] = WMMA_BF16(a[mi], bm[nt], acc[mi * 4 + nt]);
}

template <int MODE>
__global__ __launch_bounds__(32) void proj_gemm_kernel(const bf16_t* __restrict__ X,
                                                       const bf16_t* __restrict__ WT,
                                                       const float* __restrict__ bias,
                                                       bf16_t* __restrict__ out_bf,
                                                       float* __restrict__ out_f) {
    const int K = E_DIM;
    int lane = threadIdx.x & 31;
    int hi = lane >> 4;
    int ln = lane & 15;
    int m0 = blockIdx.x * 64;
    int n0 = blockIdx.y * 64;

    const bf16_t* xr[4];
    const bf16_t* wr[4];
#pragma unroll
    for (int mi = 0; mi < 4; ++mi) xr[mi] = X + (m0 + mi * 16 + ln) * K;
#pragma unroll
    for (int nt = 0; nt < 4; ++nt) wr[nt] = WT + (n0 + nt * 16 + ln) * K;

    v8f acc[16] = {};   // acc[mi*4 + nt]

    v16bf aA[4], bA[4], aB[4], bB[4];
    gemm_load(xr, wr, 0, hi, aA, bA);

#pragma unroll
    for (int kb = 0; kb < K; kb += 64) {
        gemm_load(xr, wr, kb + 32, hi, aB, bB);             // kb+32 always < K
        gemm_mma(aA, bA, acc);
        int k2 = (kb + 64 < K) ? (kb + 64) : 0;             // constant-folded per unrolled iter
        gemm_load(xr, wr, k2, hi, aA, bA);
        gemm_mma(aB, bB, acc);
    }

#pragma unroll
    for (int mi = 0; mi < 4; ++mi) {
#pragma unroll
        for (int nt = 0; nt < 4; ++nt) {
            int n = n0 + nt * 16 + ln;
            float bn = bias[n];
#pragma unroll
            for (int r = 0; r < 8; ++r) {
                int m = m0 + mi * 16 + r + hi * 8;
                float val = acc[mi * 4 + nt][r] + bn;
                if (MODE == 2) {
                    out_f[m * E_DIM + n] = val;
                } else {
                    int bb = m / T_DIM;
                    int t  = m % T_DIM;
                    int h  = n >> 6;
                    int d  = n & 63;
                    int idx = (MODE == 0)
                        ? (((bb * H_DIM + h) * T_DIM + t) * D_DIM + d)
                        : (((bb * H_DIM + h) * D_DIM + d) * T_DIM + t);
                    out_bf[idx] = (bf16_t)val;
                }
            }
        }
    }
}

// ---------------- flash attention ----------------
// One wave per (batch, head, 32-query block): two 16-row query tiles share
// every K/V fragment (halves K/V traffic per WMMA). Keys stream 32 at a time.
// q0 is 32-aligned, so exactly one (uniform) partial diagonal chunk exists.
// K fragments ping-pong between two register sets via manual 2x unroll.

__device__ __forceinline__ void attn_chunk(int jb, int jb_last, int q0, int hi, int ln,
                                           const bf16_t* __restrict__ Kh,
                                           const bf16_t* __restrict__ Vh,
                                           const v16bf (&qa)[2][2],
                                           v16bf (&kcur)[4], v16bf (&knxt)[4],
                                           v8f (&acc)[2][4],
                                           float (&mrow)[2][8], float (&lrow)[2][8],
                                           bf16_t* Plds) {
    const float scale = 0.125f;   // 1/sqrt(64)

    // S = Q @ K^T : per q-tile, two 16x16 key tiles
    v8f s[2][2] = {};
#pragma unroll
    for (int qi = 0; qi < 2; ++qi) {
        s[qi][0] = WMMA_BF16(qa[qi][0], kcur[0], s[qi][0]);
        s[qi][0] = WMMA_BF16(qa[qi][1], kcur[1], s[qi][0]);
        s[qi][1] = WMMA_BF16(qa[qi][0], kcur[2], s[qi][1]);
        s[qi][1] = WMMA_BF16(qa[qi][1], kcur[3], s[qi][1]);
    }

    // issue next-chunk K loads and current-chunk V loads now (overlap softmax)
    int jn = (jb + 32 <= jb_last) ? (jb + 32) : jb;
#pragma unroll
    for (int t = 0; t < 2; ++t) {
        const bf16_t* krow = Kh + (jn + t * 16 + ln) * D_DIM;
        knxt[t * 2 + 0] = *(const v16bf*)(krow + hi * 16);
        knxt[t * 2 + 1] = *(const v16bf*)(krow + 32 + hi * 16);
    }
    v16bf vbf[4];
#pragma unroll
    for (int dt = 0; dt < 4; ++dt)
        vbf[dt] = *(const v16bf*)(Vh + (dt * 16 + ln) * T_DIM + jb + hi * 16);
    __builtin_prefetch(Vh + jb + 32, 0, 1);

    const bool partial = (jb == jb_last);

#pragma unroll
    for (int qi = 0; qi < 2; ++qi) {
#pragma unroll
        for (int r = 0; r < 8; ++r) {
            float s0 = s[qi][0][r] * scale;
            float s1 = s[qi][1][r] * scale;
            if (partial) {
                int row = q0 + qi * 16 + r + hi * 8;
                if (jb + ln > row)      s0 = -1e30f;
                if (jb + 16 + ln > row) s1 = -1e30f;
            }
            float mx    = rowmax16(fmaxf(s0, s1));
            float mnew  = fmaxf(mrow[qi][r], mx);
            float alpha = __expf(mrow[qi][r] - mnew);
            mrow[qi][r] = mnew;
            float p0 = __expf(s0 - mnew);
            float p1 = __expf(s1 - mnew);
            lrow[qi][r] = lrow[qi][r] * alpha + rowsum16(p0 + p1);
#pragma unroll
            for (int dt = 0; dt < 4; ++dt) acc[qi][dt][r] *= alpha;
            int row_l = qi * 16 + r + hi * 8;
            Plds[row_l * 32 + ln]      = (bf16_t)p0;
            Plds[row_l * 32 + 16 + ln] = (bf16_t)p1;
        }
    }

    // O += P @ V. Single wave: LDS ops in-order; compiler inserts s_wait_dscnt
    // for the store->load dependency.
#pragma unroll
    for (int qi = 0; qi < 2; ++qi) {
        const bf16_t* prow = &Plds[(qi * 16 + ln) * 32];
        v16bf pa = make_afrag(prow + hi * 16, prow + 8 + hi * 16);
#pragma unroll
        for (int dt = 0; dt < 4; ++dt)
            acc[qi][dt] = WMMA_BF16(pa, vbf[dt], acc[qi][dt]);
    }
}

__global__ __launch_bounds__(32) void attn_kernel(const bf16_t* __restrict__ Q,
                                                  const bf16_t* __restrict__ Kc,
                                                  const bf16_t* __restrict__ Vt,
                                                  bf16_t* __restrict__ O) {
    __shared__ bf16_t Plds[32 * 32];

    int lane = threadIdx.x & 31;
    int hi = lane >> 4;
    int ln = lane & 15;

    const int nqb = T_DIM / 32;   // 64
    int qblk = blockIdx.x % nqb;
    int h    = (blockIdx.x / nqb) % H_DIM;
    int bb   = blockIdx.x / (nqb * H_DIM);
    int q0   = qblk * 32;

    const bf16_t* Qh = Q  + ((bb * H_DIM + h) * T_DIM) * D_DIM;
    const bf16_t* Kh = Kc + ((bb * H_DIM + h) * T_DIM) * D_DIM;
    const bf16_t* Vh = Vt + ((bb * H_DIM + h) * D_DIM) * T_DIM;

    // Q fragments: qa[qi][0] covers d=0..31, qa[qi][1] covers d=32..63
    v16bf qa[2][2];
#pragma unroll
    for (int qi = 0; qi < 2; ++qi) {
        const bf16_t* qrow = Qh + (q0 + qi * 16 + ln) * D_DIM;
        qa[qi][0] = make_afrag(qrow + hi * 16,      qrow + hi * 16 + 8);
        qa[qi][1] = make_afrag(qrow + 32 + hi * 16, qrow + 32 + hi * 16 + 8);
    }

    v8f acc[2][4] = {};
    float mrow[2][8], lrow[2][8];
#pragma unroll
    for (int qi = 0; qi < 2; ++qi)
#pragma unroll
        for (int r = 0; r < 8; ++r) { mrow[qi][r] = -1e30f; lrow[qi][r] = 0.0f; }

    const int jb_last = q0;   // q0 is 32-aligned: diagonal chunk starts at q0

    // preload K fragments for jb = 0 into set A: k[t*2 + dhalf]
    v16bf kA[4], kB[4];
#pragma unroll
    for (int t = 0; t < 2; ++t) {
        const bf16_t* krow = Kh + (t * 16 + ln) * D_DIM;
        kA[t * 2 + 0] = *(const v16bf*)(krow + hi * 16);
        kA[t * 2 + 1] = *(const v16bf*)(krow + 32 + hi * 16);
    }

    int jb = 0;
    for (;;) {
        attn_chunk(jb, jb_last, q0, hi, ln, Kh, Vh, qa, kA, kB, acc, mrow, lrow, Plds);
        jb += 32;
        if (jb > jb_last) break;
        attn_chunk(jb, jb_last, q0, hi, ln, Kh, Vh, qa, kB, kA, acc, mrow, lrow, Plds);
        jb += 32;
        if (jb > jb_last) break;
    }

    // normalize and write concat-head layout [B*T, E] as bf16
#pragma unroll
    for (int qi = 0; qi < 2; ++qi) {
#pragma unroll
        for (int r = 0; r < 8; ++r) {
            float inv = 1.0f / lrow[qi][r];
            int tok = bb * T_DIM + q0 + qi * 16 + r + hi * 8;
#pragma unroll
            for (int dt = 0; dt < 4; ++dt) {
                O[tok * E_DIM + h * D_DIM + dt * 16 + ln] = (bf16_t)(acc[qi][dt][r] * inv);
            }
        }
    }
}

// ---------------- host launch ----------------

extern "C" void kernel_launch(void* const* d_in, const int* in_sizes, int n_in,
                              void* d_out, int out_size, void* d_ws, size_t ws_size,
                              hipStream_t stream) {
    const float* x  = (const float*)d_in[0];
    const float* Wq = (const float*)d_in[1];
    const float* bq = (const float*)d_in[2];
    const float* Wk = (const float*)d_in[3];
    const float* bk = (const float*)d_in[4];
    const float* Wv = (const float*)d_in[5];
    const float* bv = (const float*)d_in[6];
    const float* Wo = (const float*)d_in[7];
    const float* bo = (const float*)d_in[8];
    float* out = (float*)d_out;

    char* ws = (char*)d_ws;
    size_t off = 0;
    auto take = [&](size_t bytes) -> void* {
        void* p = ws + off;
        off += (bytes + 255) & ~(size_t)255;
        return p;
    };

    bf16_t* Xbf   = (bf16_t*)take((size_t)M_DIM * E_DIM * 2);
    bf16_t* WqT   = (bf16_t*)take((size_t)E_DIM * E_DIM * 2);
    bf16_t* WkT   = (bf16_t*)take((size_t)E_DIM * E_DIM * 2);
    bf16_t* WvT   = (bf16_t*)take((size_t)E_DIM * E_DIM * 2);
    bf16_t* WoT   = (bf16_t*)take((size_t)E_DIM * E_DIM * 2);
    bf16_t* Qbuf  = (bf16_t*)take((size_t)M_DIM * E_DIM * 2);
    bf16_t* Kbuf  = (bf16_t*)take((size_t)M_DIM * E_DIM * 2);
    bf16_t* Vtbuf = (bf16_t*)take((size_t)M_DIM * E_DIM * 2);
    bf16_t* Abuf  = (bf16_t*)take((size_t)M_DIM * E_DIM * 2);  // attention output, concat heads

    // 1) conversions
    {
        int n4 = (M_DIM * E_DIM) / 4;
        f32_to_bf16_x4_kernel<<<(n4 + 255) / 256, 256, 0, stream>>>(x, Xbf, n4);
        dim3 tb(32, 8);
        dim3 tg(E_DIM / 32, E_DIM / 32);
        transpose_w_bf16_kernel<<<tg, tb, 0, stream>>>(Wq, WqT);
        transpose_w_bf16_kernel<<<tg, tb, 0, stream>>>(Wk, WkT);
        transpose_w_bf16_kernel<<<tg, tb, 0, stream>>>(Wv, WvT);
        transpose_w_bf16_kernel<<<tg, tb, 0, stream>>>(Wo, WoT);
    }

    // 2) Q/K/V projections (64x64 tile per wave)
    {
        dim3 g(M_DIM / 64, E_DIM / 64);
        proj_gemm_kernel<0><<<g, 32, 0, stream>>>(Xbf, WqT, bq, Qbuf,  nullptr);
        proj_gemm_kernel<0><<<g, 32, 0, stream>>>(Xbf, WkT, bk, Kbuf,  nullptr);
        proj_gemm_kernel<1><<<g, 32, 0, stream>>>(Xbf, WvT, bv, Vtbuf, nullptr);
    }

    // 3) causal flash attention (32 queries per wave)
    {
        int nblk = B_DIM * H_DIM * (T_DIM / 32);   // 2048 waves
        attn_kernel<<<nblk, 32, 0, stream>>>(Qbuf, Kbuf, Vtbuf, Abuf);
    }

    // 4) output projection (fp32 result + bias)
    {
        dim3 g(M_DIM / 64, E_DIM / 64);
        proj_gemm_kernel<2><<<g, 32, 0, stream>>>(Abuf, WoT, bo, nullptr, out);
    }

    (void)in_sizes; (void)n_in; (void)out_size; (void)ws_size;
}
